// TextDecoder_40733469835780
// MI455X (gfx1250) — matrix operand order repaired and checked by reading, not compiled, not run on hardware
//
#include <hip/hip_runtime.h>

// ---------------------------------------------------------------------------
// Whisper text-decoder forward for MI455X (gfx1250, wave32, WMMA).
// All dense math runs through one batched/strided WMMA GEMM kernel using
// v_wmma_f32_16x16x32_f16 (f32 accumulate, f16 operands staged in LDS).
// 128x64 block tile, 8 waves * (32x32 per wave -> 4 WMMAs), double-buffered LDS.
// ---------------------------------------------------------------------------

typedef __attribute__((ext_vector_type(16))) _Float16 v16h;
typedef __attribute__((ext_vector_type(8)))  _Float16 v8h;
typedef __attribute__((ext_vector_type(8)))  float    v8f;
typedef __attribute__((ext_vector_type(4)))  float    v4f;

#define L_    12
#define D_    768
#define H_    12
#define V_    51865
#define B_    2
#define T_    192
#define MAXC_ 256
#define DH_   64
#define SA_   1500

#define BM 128
#define BN 64
#define BK 32
#define LDSP (BK + 8)   // pad LDS rows (80B, multiple of 16B) to soften bank conflicts

struct GemmParams {
  const float* A; const float* W; const float* bias; float* C;
  int M, N, K, nH;                 // batch z -> (z / nH, z % nH)
  int sAm, sAk, sAbh, sAbl;        // A: row stride, k stride, hi/lo batch strides
  int sWn, sWk, sWbh, sWbl;        // W: row(n) stride, k stride, hi/lo batch strides
  int sCm, sCbh, sCbl;             // C: row stride (col stride == 1), batch strides
};

__device__ __forceinline__ v16h cat16(v8h lo, v8h hi) {
  return __builtin_shufflevector(lo, hi, 0,1,2,3,4,5,6,7,8,9,10,11,12,13,14,15);
}

__device__ __forceinline__ v8h cvt8(v4f a0, v4f a1) {
  v8h h;
  h[0] = (_Float16)a0[0]; h[1] = (_Float16)a0[1];
  h[2] = (_Float16)a0[2]; h[3] = (_Float16)a0[3];
  h[4] = (_Float16)a1[0]; h[5] = (_Float16)a1[1];
  h[6] = (_Float16)a1[2]; h[7] = (_Float16)a1[3];
  return h;
}

__device__ __forceinline__ v8f wmma_f16(v16h a, v16h b, v8f c) {
  return __builtin_amdgcn_wmma_f32_16x16x32_f16(false, a, false, b,
                                                (short)0, c, false, false);
}

// 4 WMMAs: wave-owned 32x32 C sub-tile from a 32-row A strip and 32-col B strip.
__device__ __forceinline__ void mma_step4(const _Float16 (*As)[LDSP],
                                          const _Float16 (*Bs)[LDSP],
                                          int wm, int wn, int lane,
                                          v8f& c00, v8f& c01, v8f& c10, v8f& c11) {
  const int r15 = lane & 15;
  const int kbA = (lane & 16) ? 8 : 0;    // A frag: split-K half pattern
  const int kbB = (lane & 16) ? 8 : 0;    // B frag: split-K half pattern
  v16h af0 = cat16(*(const v8h*)&As[wm + r15][kbA],
                   *(const v8h*)&As[wm + r15][kbA + 16]);
  v16h af1 = cat16(*(const v8h*)&As[wm + 16 + r15][kbA],
                   *(const v8h*)&As[wm + 16 + r15][kbA + 16]);
  v16h bf0 = cat16(*(const v8h*)&Bs[wn + r15][kbB],
                   *(const v8h*)&Bs[wn + r15][kbB + 16]);
  v16h bf1 = cat16(*(const v8h*)&Bs[wn + 16 + r15][kbB],
                   *(const v8h*)&Bs[wn + 16 + r15][kbB + 16]);
  c00 = wmma_f16(af0, bf0, c00);
  c01 = wmma_f16(af0, bf1, c01);
  c10 = wmma_f16(af1, bf0, c10);
  c11 = wmma_f16(af1, bf1, c11);
}

// C[m,n] = sum_k A[m,k] * W[n,k] (+ bias[n]); strided+batched; f16 WMMA, f32 acc.
__global__ __launch_bounds__(256) void gemm_wmma(GemmParams p) {
  __shared__ __align__(16) _Float16 As[2][BM][LDSP];
  __shared__ __align__(16) _Float16 Bs[2][BN][LDSP];

  const int z  = blockIdx.z;
  const int zb = z / p.nH, zh = z % p.nH;
  const float* __restrict__ A = p.A + zb * p.sAbh + zh * p.sAbl;
  const float* __restrict__ W = p.W + zb * p.sWbh + zh * p.sWbl;
  float*       __restrict__ C = p.C + zb * p.sCbh + zh * p.sCbl;

  const int m0   = blockIdx.y * BM;
  const int n0   = blockIdx.x * BN;
  const int tid  = threadIdx.x;
  const int lane = tid & 31;
  const int wave = tid >> 5;           // 8 waves: 4 (M) x 2 (N)
  const int wm   = (wave >> 1) << 5;   // 0,32,64,96
  const int wn   = (wave & 1) << 5;    // 0,32

  // cooperative-load mapping: A = 128 rows x 32 k (16 halves/thread),
  //                           B =  64 rows x 32 k ( 8 halves/thread)
  const int rA = tid >> 1, kA = (tid & 1) << 4;   // rows 0..127, k 0/16
  const int rB = tid >> 2, kB = (tid & 3) << 3;   // rows 0..63,  k 0,8,16,24

  // Clamp OOB rows to a valid address: their LDS data only feeds C rows/cols
  // that are never stored, so no zeroing (and no divergent guard) is needed.
  const int ar = (m0 + rA < p.M) ? (m0 + rA) : (p.M - 1);
  const int br = (n0 + rB < p.N) ? (n0 + rB) : (p.N - 1);
  const float* __restrict__ Arow = A + ar * p.sAm;
  const float* __restrict__ Wrow = W + br * p.sWn;
  const bool vecA = (p.sAk == 1);
  const bool vecB = (p.sWk == 1);

  v8f acc00 = {}, acc01 = {}, acc10 = {}, acc11 = {};
  const int kFull = p.K & ~(BK - 1);

  v8h a0, a1, b0;                      // register-staged next tile (f16)
  auto stage = [&](int kk) {           // global -> regs (+f32->f16 convert)
    if (vecA) {
      const v4f* pa = (const v4f*)(Arow + kk + kA);
      a0 = cvt8(pa[0], pa[1]);
      a1 = cvt8(pa[2], pa[3]);
    } else {
      v8h t0, t1;
#pragma unroll
      for (int u = 0; u < 8; ++u) {
        t0[u] = (_Float16)Arow[(kk + kA + u) * p.sAk];
        t1[u] = (_Float16)Arow[(kk + kA + 8 + u) * p.sAk];
      }
      a0 = t0; a1 = t1;
    }
    if (vecB) {
      const v4f* pb = (const v4f*)(Wrow + kk + kB);
      b0 = cvt8(pb[0], pb[1]);
    } else {
      v8h t;
#pragma unroll
      for (int u = 0; u < 8; ++u)
        t[u] = (_Float16)Wrow[(kk + kB + u) * p.sWk];
      b0 = t;
    }
  };
  auto commit = [&](int buf) {         // regs -> LDS (ds_store_b128 x3)
    *(v8h*)&As[buf][rA][kA]     = a0;
    *(v8h*)&As[buf][rA][kA + 8] = a1;
    *(v8h*)&Bs[buf][rB][kB]     = b0;
  };

  int cur = 0;
  if (kFull > 0) {                     // prologue: tile 0 into buffer 0
    stage(0);
    commit(0);
    __syncthreads();
  }
  for (int k0 = 0; k0 < kFull; k0 += BK) {
    const int  kn      = k0 + BK;
    const bool hasNext = kn < kFull;
    if (hasNext) {                     // issue next-tile loads before the MMAs
      stage(kn);
      if (kn + BK < kFull) {
        if (vecA) __builtin_prefetch(Arow + kn + BK + kA, 0, 1);
        if (vecB) __builtin_prefetch(Wrow + kn + BK + kB, 0, 1);
      }
    }
    mma_step4(As[cur], Bs[cur], wm, wn, lane, acc00, acc01, acc10, acc11);
    if (hasNext) commit(cur ^ 1);
    __syncthreads();
    cur ^= 1;
  }

  if (kFull < p.K) {                   // single guarded K-tail step
    v8h t0, t1, t2;
#pragma unroll
    for (int u = 0; u < 8; ++u) {
      const int ka = kFull + kA + u, kb2 = ka + 8, kw = kFull + kB + u;
      const int kca = (ka  < p.K) ? ka  : (p.K - 1);
      const int kcb = (kb2 < p.K) ? kb2 : (p.K - 1);
      const int kcw = (kw  < p.K) ? kw  : (p.K - 1);
      const float va = Arow[kca * p.sAk];
      const float vb = Arow[kcb * p.sAk];
      const float vw = Wrow[kcw * p.sWk];
      t0[u] = (_Float16)((ka  < p.K) ? va : 0.f);
      t1[u] = (_Float16)((kb2 < p.K) ? vb : 0.f);
      t2[u] = (_Float16)((kw  < p.K) ? vw : 0.f);
    }
    a0 = t0; a1 = t1; b0 = t2;
    commit(cur);
    __syncthreads();
    mma_step4(As[cur], Bs[cur], wm, wn, lane, acc00, acc01, acc10, acc11);
  }

  // D layout: lane = N(+16*hi), VGPR r -> M = r + 8*hi
  const int n  = lane & 15;
  const int mb = (lane >> 4) << 3;
  const int c0 = n0 + wn + n;
  const int c1 = c0 + 16;
  const float bv0 = (p.bias && c0 < p.N) ? p.bias[c0] : 0.f;
  const float bv1 = (p.bias && c1 < p.N) ? p.bias[c1] : 0.f;
#pragma unroll
  for (int r = 0; r < 8; ++r) {
    const int m  = m0 + wm + mb + r;
    const int m2 = m + 16;
    if (m < p.M) {
      if (c0 < p.N) C[m * p.sCm + c0] = acc00[r] + bv0;
      if (c1 < p.N) C[m * p.sCm + c1] = acc01[r] + bv1;
    }
    if (m2 < p.M) {
      if (c0 < p.N) C[m2 * p.sCm + c0] = acc10[r] + bv0;
      if (c1 < p.N) C[m2 * p.sCm + c1] = acc11[r] + bv1;
    }
  }
}

// h[b,t,:] = tok_emb[x[b,t]] + pos_emb[t]  (t < T), else 0
__global__ __launch_bounds__(256) void embed_k(const int* __restrict__ x,
                                               const float* __restrict__ tok,
                                               const float* __restrict__ pos,
                                               float* __restrict__ h) {
  const int row = blockIdx.x;          // b*256 + t
  const int b = row >> 8, t = row & 255;
  float* hr = h + (long long)row * D_;
  if (t < T_) {
    const float* tr = tok + (long long)x[b * T_ + t] * D_;
    const float* pr = pos + (long long)t * D_;
#pragma unroll
    for (int u = 0; u < 3; ++u) { int d = threadIdx.x + u * 256; hr[d] = tr[d] + pr[d]; }
  } else {
#pragma unroll
    for (int u = 0; u < 3; ++u) { int d = threadIdx.x + u * 256; hr[d] = 0.f; }
  }
}

__global__ __launch_bounds__(256) void layernorm_k(const float* __restrict__ x,
                                                   const float* __restrict__ g,
                                                   const float* __restrict__ b,
                                                   float* __restrict__ y) {
  __shared__ float r1[256], r2[256];
  const int row = blockIdx.x;
  const int t = threadIdx.x;
  const float* xr = x + (long long)row * D_;
  float v0 = xr[t], v1 = xr[t + 256], v2 = xr[t + 512];
  r1[t] = v0 + v1 + v2;
  r2[t] = v0 * v0 + v1 * v1 + v2 * v2;
  __syncthreads();
  for (int off = 128; off > 0; off >>= 1) {
    if (t < off) { r1[t] += r1[t + off]; r2[t] += r2[t + off]; }
    __syncthreads();
  }
  const float mean = r1[0] * (1.f / (float)D_);
  const float var  = r2[0] * (1.f / (float)D_) - mean * mean;
  const float inv  = rsqrtf(var + 1e-5f);
  float* yr = y + (long long)row * D_;
  yr[t]       = (v0 - mean) * inv * g[t]       + b[t];
  yr[t + 256] = (v1 - mean) * inv * g[t + 256] + b[t + 256];
  yr[t + 512] = (v2 - mean) * inv * g[t + 512] + b[t + 512];
}

// self-attention: in-place softmax over 256 keys with causal + T mask, after *0.125
__global__ __launch_bounds__(256) void softmax_self_k(float* __restrict__ s) {
  __shared__ float red[256];
  const int row = blockIdx.x;          // ((b*H+h)*256 + i)
  const int i = row & 255;
  const int j = threadIdx.x;
  float* p = s + (long long)row * 256;
  const bool masked = (j > i) || (j >= T_);
  const float v = masked ? -__builtin_inff() : p[j] * 0.125f;
  red[j] = v; __syncthreads();
  for (int off = 128; off > 0; off >>= 1) {
    if (j < off) red[j] = fmaxf(red[j], red[j + off]);
    __syncthreads();
  }
  const float mx = red[0]; __syncthreads();
  const float e = masked ? 0.f : __expf(v - mx);
  red[j] = e; __syncthreads();
  for (int off = 128; off > 0; off >>= 1) {
    if (j < off) red[j] += red[j + off];
    __syncthreads();
  }
  p[j] = e / red[0];
}

// cross-attention: emit scaled raw scores (rows < T) to cross_qks, softmax in place
__global__ __launch_bounds__(256) void softmax_cross_k(float* __restrict__ s,
                                                       float* __restrict__ cqk, int l) {
  __shared__ float red[256];
  const int row = blockIdx.x;          // (b*H+h)*256 + i
  const int i = row & 255;
  const int bh = row >> 8;
  const int b = bh / H_, h = bh % H_;
  const int t = threadIdx.x;
  float* p = s + (long long)row * SA_;
  float vals[6];
  float mx = -__builtin_inff();
#pragma unroll
  for (int u = 0; u < 6; ++u) {
    const int j = t + u * 256;
    vals[u] = (j < SA_) ? p[j] * 0.125f : -__builtin_inff();
    mx = fmaxf(mx, vals[u]);
  }
  red[t] = mx; __syncthreads();
  for (int off = 128; off > 0; off >>= 1) {
    if (t < off) red[t] = fmaxf(red[t], red[t + off]);
    __syncthreads();
  }
  mx = red[0]; __syncthreads();
  float es[6]; float sum = 0.f;
#pragma unroll
  for (int u = 0; u < 6; ++u) {
    const int j = t + u * 256;
    es[u] = (j < SA_) ? __expf(vals[u] - mx) : 0.f;
    sum += es[u];
  }
  red[t] = sum; __syncthreads();
  for (int off = 128; off > 0; off >>= 1) {
    if (t < off) red[t] += red[t + off];
    __syncthreads();
  }
  const float rinv = 1.f / red[0];
  if (i < T_) {
    float* op = cqk + ((long long)((l * B_ + b) * H_ + h) * T_ + i) * SA_;
#pragma unroll
    for (int u = 0; u < 6; ++u) { const int j = t + u * 256; if (j < SA_) op[j] = vals[u]; }
  }
#pragma unroll
  for (int u = 0; u < 6; ++u) { const int j = t + u * 256; if (j < SA_) p[j] = es[u] * rinv; }
}

__global__ __launch_bounds__(256) void gelu_k(float* __restrict__ p) {
  const int i = blockIdx.x * 256 + threadIdx.x;
  const float x = p[i];
  p[i] = 0.5f * x * (1.f + erff(x * 0.70710678118654752f));
}

__global__ __launch_bounds__(256) void addv_k(float* __restrict__ h, const float* __restrict__ s) {
  const int i = blockIdx.x * 256 + threadIdx.x;
  h[i] += s[i];
}

// ---------------------------------------------------------------------------
static inline void run_gemm(hipStream_t st, const float* A, const float* W,
                            const float* bias, float* C,
                            int M, int N, int K, int batch, int nH,
                            int sAm, int sAk, int sAbh, int sAbl,
                            int sWn, int sWk, int sWbh, int sWbl,
                            int sCm, int sCbh, int sCbl) {
  GemmParams p;
  p.A = A; p.W = W; p.bias = bias; p.C = C;
  p.M = M; p.N = N; p.K = K; p.nH = nH;
  p.sAm = sAm; p.sAk = sAk; p.sAbh = sAbh; p.sAbl = sAbl;
  p.sWn = sWn; p.sWk = sWk; p.sWbh = sWbh; p.sWbl = sWbl;
  p.sCm = sCm; p.sCbh = sCbh; p.sCbl = sCbl;
  dim3 grid((N + BN - 1) / BN, (M + BM - 1) / BM, batch);
  gemm_wmma<<<grid, 256, 0, st>>>(p);
}

extern "C" void kernel_launch(void* const* d_in, const int* in_sizes, int n_in,
                              void* d_out, int out_size, void* d_ws, size_t ws_size,
                              hipStream_t stream) {
  (void)in_sizes; (void)n_in; (void)out_size; (void)ws_size;

  const int*   x       = (const int*)  d_in[0];
  const float* xa      = (const float*)d_in[1];
  const float* tok     = (const float*)d_in[4];
  const float* pos     = (const float*)d_in[5];
  const float* a_ln_g  = (const float*)d_in[6];
  const float* a_ln_b  = (const float*)d_in[7];
  const float* a_q_w   = (const float*)d_in[8];
  const float* a_q_b   = (const float*)d_in[9];
  const float* a_k_w   = (const float*)d_in[10];
  const float* a_v_w   = (const float*)d_in[11];
  const float* a_v_b   = (const float*)d_in[12];
  const float* a_o_w   = (const float*)d_in[13];
  const float* a_o_b   = (const float*)d_in[14];
  const float* c_ln_g  = (const float*)d_in[15];
  const float* c_ln_b  = (const float*)d_in[16];
  const float* c_q_w   = (const float*)d_in[17];
  const float* c_q_b   = (const float*)d_in[18];
  const float* c_k_w   = (const float*)d_in[19];
  const float* c_v_w   = (const float*)d_in[20];
  const float* c_v_b   = (const float*)d_in[21];
  const float* c_o_w   = (const float*)d_in[22];
  const float* c_o_b   = (const float*)d_in[23];
  const float* m_ln_g  = (const float*)d_in[24];
  const float* m_ln_b  = (const float*)d_in[25];
  const float* m_w1    = (const float*)d_in[26];
  const float* m_b1    = (const float*)d_in[27];
  const float* m_w2    = (const float*)d_in[28];
  const float* m_b2    = (const float*)d_in[29];
  const float* f_ln_g  = (const float*)d_in[30];
  const float* f_ln_b  = (const float*)d_in[31];

  float* out = (float*)d_out;
  // output layout: logits | cross_qks | new_mkv | new_ckv
  float* out_logits = out;                                   // B*T*V
  float* out_cqk    = out + 19916160LL;                      // L*B * H * T * 1500
  float* out_mkv    = out + 102860160LL;                     // 24 * B*256*768
  float* out_ckv    = out + 112297344LL;                     // 24 * 1500*768

  // workspace carve (floats)
  float* ws  = (float*)d_ws;
  const long long BT = (long long)B_ * MAXC_ * D_;           // 393216
  float* h   = ws;
  float* y   = h   + BT;
  float* q   = y   + BT;
  float* o   = q   + BT;
  float* mid = o   + BT;                                     // B*256*3072
  float* sc  = mid + (long long)B_ * MAXC_ * 4 * D_;         // B*H*256*1500

  const long long DD  = (long long)D_ * D_;
  const long long KVL = (long long)SA_ * D_;
  const int BH = B_ * H_;

  embed_k<<<B_ * MAXC_, 256, 0, stream>>>(x, tok, pos, h);

  for (int l = 0; l < L_; ++l) {
    float* kv_k = out_mkv + (2LL * l)     * BT;
    float* kv_v = out_mkv + (2LL * l + 1) * BT;
    float* ck   = out_ckv + (2LL * l)     * KVL;
    float* cv   = out_ckv + (2LL * l + 1) * KVL;

    // ----- self-attention -----
    layernorm_k<<<B_ * MAXC_, 256, 0, stream>>>(h, a_ln_g + l * D_, a_ln_b + l * D_, y);
    run_gemm(stream, y, a_q_w + l * DD, a_q_b + l * D_, q,    512, D_, D_, 1, 1,
             D_, 1, 0, 0,   D_, 1, 0, 0,   D_, 0, 0);
    run_gemm(stream, y, a_k_w + l * DD, nullptr,        kv_k, 512, D_, D_, 1, 1,
             D_, 1, 0, 0,   D_, 1, 0, 0,   D_, 0, 0);
    run_gemm(stream, y, a_v_w + l * DD, a_v_b + l * D_, kv_v, 512, D_, D_, 1, 1,
             D_, 1, 0, 0,   D_, 1, 0, 0,   D_, 0, 0);
    // scores[b,h] = q_h @ k_h^T   (M=256, N=256, K=64)
    run_gemm(stream, q, kv_k, nullptr, sc, 256, 256, 64, BH, H_,
             D_, 1, 196608, 64,   D_, 1, 196608, 64,   256, 786432, 65536);
    softmax_self_k<<<BH * 256, 256, 0, stream>>>(sc);
    // o[b,h] = probs @ v_h   (W = v^T via k-stride 768)
    run_gemm(stream, sc, kv_v, nullptr, o, 256, 64, 256, BH, H_,
             256, 1, 786432, 65536,   1, D_, 196608, 64,   D_, 196608, 64);
    run_gemm(stream, o, a_o_w + l * DD, a_o_b + l * D_, y, 512, D_, D_, 1, 1,
             D_, 1, 0, 0,   D_, 1, 0, 0,   D_, 0, 0);
    addv_k<<<1536, 256, 0, stream>>>(h, y);

    // ----- cross-attention -----
    layernorm_k<<<B_ * MAXC_, 256, 0, stream>>>(h, c_ln_g + l * D_, c_ln_b + l * D_, y);
    run_gemm(stream, y, c_q_w + l * DD, c_q_b + l * D_, q, 512, D_, D_, 1, 1,
             D_, 1, 0, 0,   D_, 1, 0, 0,   D_, 0, 0);
    // ck/cv from xa[:1] (batch 0 only), written straight into new_ckv output
    run_gemm(stream, xa, c_k_w + l * DD, nullptr,        ck, SA_, D_, D_, 1, 1,
             D_, 1, 0, 0,   D_, 1, 0, 0,   D_, 0, 0);
    run_gemm(stream, xa, c_v_w + l * DD, c_v_b + l * D_, cv, SA_, D_, D_, 1, 1,
             D_, 1, 0, 0,   D_, 1, 0, 0,   D_, 0, 0);
    // scores[b,h] = q_h @ ck_h^T  (ck has no b dependence -> sWbh = 0)
    run_gemm(stream, q, ck, nullptr, sc, 256, SA_, 64, BH, H_,
             D_, 1, 196608, 64,   D_, 1, 0, 64,   SA_, 4608000, 384000);
    softmax_cross_k<<<BH * 256, 256, 0, stream>>>(sc, out_cqk, l);
    run_gemm(stream, sc, cv, nullptr, o, 256, 64, SA_, BH, H_,
             SA_, 1, 4608000, 384000,   1, D_, 0, 64,   D_, 196608, 64);
    run_gemm(stream, o, c_o_w + l * DD, c_o_b + l * D_, y, 512, D_, D_, 1, 1,
             D_, 1, 0, 0,   D_, 1, 0, 0,   D_, 0, 0);
    addv_k<<<1536, 256, 0, stream>>>(h, y);

    // ----- MLP -----
    layernorm_k<<<B_ * MAXC_, 256, 0, stream>>>(h, m_ln_g + l * D_, m_ln_b + l * D_, y);
    run_gemm(stream, y, m_w1 + (long long)l * 4 * DD, m_b1 + (long long)l * 4 * D_, mid,
             512, 4 * D_, D_, 1, 1,
             D_, 1, 0, 0,   D_, 1, 0, 0,   4 * D_, 0, 0);
    gelu_k<<<6144, 256, 0, stream>>>(mid);
    run_gemm(stream, mid, m_w2 + (long long)l * 4 * DD, m_b2 + l * D_, y,
             512, D_, 4 * D_, 1, 1,
             4 * D_, 1, 0, 0,   4 * D_, 1, 0, 0,   D_, 0, 0);
    addv_k<<<1536, 256, 0, stream>>>(h, y);
  }

  // final LN + logits = h[:, :T] @ tok_emb^T  (per-batch rows 0..191 are contiguous)
  layernorm_k<<<B_ * MAXC_, 256, 0, stream>>>(h, f_ln_g, f_ln_b, y);
  run_gemm(stream, y, tok, nullptr, out_logits, T_, V_, D_, B_, 1,
           D_, 1, 196608, 0,   D_, 1, 0, 0,   V_, T_ * V_, 0);
}
